// Nystrom_68556267978952
// MI455X (gfx1250) — matrix operand order repaired and checked by reading, not compiled, not run on hardware
//
#include <hip/hip_runtime.h>

// ---------------- types ----------------
typedef __attribute__((ext_vector_type(16))) __bf16 bf16x16;
typedef __attribute__((ext_vector_type(8)))  __bf16 bf16x8;
typedef __attribute__((ext_vector_type(8)))  float  f32x8;
typedef __attribute__((ext_vector_type(2)))  float  f32x2;

#define M_ROWS 16384
#define DIM    1024
#define KCENT  512
#define KM_IT  10

#define LDSSTRIDE 40   // bf16 elems per LDS row (80 B) -> conflict-free quads, 16B aligned chunks

__device__ __forceinline__ void async_copy_16B(const void* gsrc, const void* ldst)
{
    const unsigned lds_off = (unsigned)(size_t)ldst;   // low 32 bits of flat LDS addr == LDS offset
    asm volatile("global_load_async_to_lds_b128 %0, %1, off"
                 :: "v"(lds_off), "v"(gsrc) : "memory");
}

__device__ __forceinline__ void wait_async0()
{
    asm volatile("s_wait_asynccnt 0x0" ::: "memory");
}

__device__ __forceinline__ bf16x16 combine8(bf16x8 lo, bf16x8 hi)
{
    return __builtin_shufflevector(lo, hi, 0,1,2,3,4,5,6,7,8,9,10,11,12,13,14,15);
}

// ---------------- bf16 GEMM: C[M,N] = A[M,K] * B[N,K]^T  (A,B bf16, C f32) ----------------
// block tile 128x64 (8 waves, wave tile 32x32), K-step 32, async double-buffered LDS staging
__global__ __launch_bounds__(256) void gemm_bf16_nt(
    const __bf16* __restrict__ A, const __bf16* __restrict__ B,
    float* __restrict__ C, int Mdim, int Ndim, int Kdim)
{
    __shared__ __attribute__((aligned(16))) __bf16 sm[2 * (128 + 64) * LDSSTRIDE];

    const int t    = threadIdx.x;
    const int lane = t & 31;
    const int wave = t >> 5;
    const int wrow = wave >> 1;          // 0..3
    const int wcol = wave & 1;           // 0..1
    const int m0 = blockIdx.y * 128;
    const int n0 = blockIdx.x * 64;
    const int lm = lane & 15;
    const int lh = lane >> 4;

    // async stage of one K-slab (A: 128x32, B: 64x32) into buffer `buf`
    auto stage = [&](int buf, int kk) {
        __bf16* smA = sm + buf * (128 + 64) * LDSSTRIDE;
        __bf16* smB = smA + 128 * LDSSTRIDE;
#pragma unroll
        for (int i = 0; i < 2; ++i) {                 // A: 512 chunks of 16B
            const int q = t + i * 256;
            const int row = q >> 2, cc = q & 3;
            async_copy_16B(A + (size_t)(m0 + row) * Kdim + kk + cc * 8,
                           smA + row * LDSSTRIDE + cc * 8);
        }
        {                                             // B: 256 chunks of 16B
            const int row = t >> 2, cc = t & 3;
            async_copy_16B(B + (size_t)(n0 + row) * Kdim + kk + cc * 8,
                           smB + row * LDSSTRIDE + cc * 8);
        }
    };

    f32x8 c00 = {}, c01 = {}, c10 = {}, c11 = {};

    const int nsteps = Kdim / 32;
    stage(0, 0);
    for (int s = 0; s < nsteps; ++s) {
        wait_async0();
        __syncthreads();                 // slab s ready in buffer s&1
        if (s + 1 < nsteps) stage((s + 1) & 1, (s + 1) * 32);

        const __bf16* smA = sm + (s & 1) * (128 + 64) * LDSSTRIDE;
        const __bf16* smB = smA + 128 * LDSSTRIDE;

        const __bf16* pa0 = smA + (wrow * 32 + lm) * LDSSTRIDE + lh * 8;
        const __bf16* pa1 = pa0 + 16 * LDSSTRIDE;
        const __bf16* pb0 = smB + (wcol * 32 + lm) * LDSSTRIDE + lh * 8;
        const __bf16* pb1 = pb0 + 16 * LDSSTRIDE;

        bf16x16 a0 = combine8(*(const bf16x8*)pa0, *(const bf16x8*)(pa0 + 16));
        bf16x16 a1 = combine8(*(const bf16x8*)pa1, *(const bf16x8*)(pa1 + 16));
        bf16x16 b0 = combine8(*(const bf16x8*)pb0, *(const bf16x8*)(pb0 + 16));
        bf16x16 b1 = combine8(*(const bf16x8*)pb1, *(const bf16x8*)(pb1 + 16));

        c00 = __builtin_amdgcn_wmma_f32_16x16x32_bf16(false, a0, false, b0, (short)0, c00, false, false);
        c01 = __builtin_amdgcn_wmma_f32_16x16x32_bf16(false, a0, false, b1, (short)0, c01, false, false);
        c10 = __builtin_amdgcn_wmma_f32_16x16x32_bf16(false, a1, false, b0, (short)0, c10, false, false);
        c11 = __builtin_amdgcn_wmma_f32_16x16x32_bf16(false, a1, false, b1, (short)0, c11, false, false);
    }

    const int mb = m0 + wrow * 32;
    const int nb = n0 + wcol * 32;
#pragma unroll
    for (int j = 0; j < 8; ++j) {
        const int r = j + 8 * lh;
        C[(size_t)(mb + r)      * Ndim + nb + lm]      = c00[j];
        C[(size_t)(mb + r)      * Ndim + nb + 16 + lm] = c01[j];
        C[(size_t)(mb + 16 + r) * Ndim + nb + lm]      = c10[j];
        C[(size_t)(mb + 16 + r) * Ndim + nb + 16 + lm] = c11[j];
    }
}

// ---------------- f32 GEMM: C[M,N] = A[M,K] * B[N,K]^T  (v_wmma_f32_16x16x4_f32) ----------------
__global__ __launch_bounds__(256) void gemm_f32_nt(
    const float* __restrict__ A, const float* __restrict__ B,
    float* __restrict__ C, int Mdim, int Ndim, int Kdim)
{
    const int lane = threadIdx.x & 31;
    const int wave = threadIdx.x >> 5;
    const int mb = blockIdx.y * 128 + (wave >> 1) * 32;
    const int nb = blockIdx.x * 64  + (wave & 1)  * 32;
    const int lm = lane & 15;
    const int lh = lane >> 4;

    f32x8 c00 = {}, c01 = {}, c10 = {}, c11 = {};

    const float* A0 = A + (size_t)(mb + lm) * Kdim;
    const float* A1 = A0 + (size_t)16 * Kdim;
    const float* B0 = B + (size_t)(nb + lm) * Kdim;
    const float* B1 = B0 + (size_t)16 * Kdim;

    for (int kk = 0; kk < Kdim; kk += 4) {
        const int ka = kk + 2 * lh;  // elems 0,1 -> K = ka, ka+1
        f32x2 a0 = *(const f32x2*)(A0 + ka);
        f32x2 a1 = *(const f32x2*)(A1 + ka);
        f32x2 b0 = *(const f32x2*)(B0 + ka);
        f32x2 b1 = *(const f32x2*)(B1 + ka);
        c00 = __builtin_amdgcn_wmma_f32_16x16x4_f32(false, a0, false, b0, (short)0, c00, false, false);
        c01 = __builtin_amdgcn_wmma_f32_16x16x4_f32(false, a0, false, b1, (short)0, c01, false, false);
        c10 = __builtin_amdgcn_wmma_f32_16x16x4_f32(false, a1, false, b0, (short)0, c10, false, false);
        c11 = __builtin_amdgcn_wmma_f32_16x16x4_f32(false, a1, false, b1, (short)0, c11, false, false);
    }
#pragma unroll
    for (int j = 0; j < 8; ++j) {
        const int r = j + 8 * lh;
        C[(size_t)(mb + r)      * Ndim + nb + lm]      = c00[j];
        C[(size_t)(mb + r)      * Ndim + nb + 16 + lm] = c01[j];
        C[(size_t)(mb + 16 + r) * Ndim + nb + lm]      = c10[j];
        C[(size_t)(mb + 16 + r) * Ndim + nb + 16 + lm] = c11[j];
    }
}

// ---------------- helpers ----------------
__global__ __launch_bounds__(256) void convert_x_bf16(const float* __restrict__ X,
                                                      __bf16* __restrict__ Xb, int n)
{
    int i = blockIdx.x * blockDim.x + threadIdx.x;
    if (i < n) Xb[i] = (__bf16)X[i];
}

__global__ __launch_bounds__(256) void init_centers(const float* __restrict__ X,
                                                    float* __restrict__ centers)
{
    const int k = blockIdx.x;           // 512 blocks
    const int src = k * (M_ROWS / KCENT);
#pragma unroll
    for (int j = 0; j < 4; ++j) {
        const int d = threadIdx.x + j * 256;
        centers[(size_t)k * DIM + d] = X[(size_t)src * DIM + d];
    }
}

// per-center: cnorm[k] = sum c^2, and bf16 copy of centers
__global__ __launch_bounds__(256) void prep_centers(const float* __restrict__ centers,
                                                    __bf16* __restrict__ Cb,
                                                    float* __restrict__ cnorm)
{
    __shared__ float red[256];
    const int k = blockIdx.x;
    float acc = 0.f;
#pragma unroll
    for (int j = 0; j < 4; ++j) {
        const int d = threadIdx.x + j * 256;
        const float v = centers[(size_t)k * DIM + d];
        Cb[(size_t)k * DIM + d] = (__bf16)v;
        acc += v * v;
    }
    red[threadIdx.x] = acc;
    __syncthreads();
    for (int s = 128; s > 0; s >>= 1) {
        if (threadIdx.x < s) red[threadIdx.x] += red[threadIdx.x + s];
        __syncthreads();
    }
    if (threadIdx.x == 0) cnorm[k] = red[0];
}

// assign[n] = argmin_k (cnorm[k] - 2*G[n][k]); one wave per row
__global__ __launch_bounds__(256) void argmin_rows(const float* __restrict__ G,
                                                   const float* __restrict__ cnorm,
                                                   int* __restrict__ assign, int cols)
{
    const int lane = threadIdx.x & 31;
    const int wave = threadIdx.x >> 5;
    const int row = blockIdx.x * 8 + wave;
    float best = 3.4e38f;
    int   bidx = 0;
    for (int k = lane; k < cols; k += 32) {
        const float v = cnorm[k] - 2.f * G[(size_t)row * cols + k];
        if (v < best || (v == best && k < bidx)) { best = v; bidx = k; }
    }
#pragma unroll
    for (int off = 16; off > 0; off >>= 1) {
        const float ov = __shfl_down(best, off, 32);
        const int   oi = __shfl_down(bidx, off, 32);
        if (ov < best || (ov == best && oi < bidx)) { best = ov; bidx = oi; }
    }
    if (lane == 0) assign[row] = bidx;
}

// deterministic segment sum: one block per center, threads own dims
__global__ __launch_bounds__(256) void seg_sum(const float* __restrict__ X,
                                               const int* __restrict__ assign,
                                               float* __restrict__ sums,
                                               float* __restrict__ cnts)
{
    const int k = blockIdx.x;
    float acc0 = 0.f, acc1 = 0.f, acc2 = 0.f, acc3 = 0.f;
    int cnt = 0;
    const int d0 = threadIdx.x;
    for (int n = 0; n < M_ROWS; ++n) {
        if (assign[n] == k) {
            const float* row = X + (size_t)n * DIM;
            acc0 += row[d0];
            acc1 += row[d0 + 256];
            acc2 += row[d0 + 512];
            acc3 += row[d0 + 768];
            ++cnt;
        }
    }
    float* s = sums + (size_t)k * DIM;
    s[d0] = acc0; s[d0 + 256] = acc1; s[d0 + 512] = acc2; s[d0 + 768] = acc3;
    if (threadIdx.x == 0) cnts[k] = (float)cnt;
}

__global__ __launch_bounds__(256) void update_centers(const float* __restrict__ sums,
                                                      const float* __restrict__ cnts,
                                                      float* __restrict__ centers)
{
    const int k = blockIdx.x;
    const float c = cnts[k];
    const float inv = 1.f / fmaxf(c, 1.f);
#pragma unroll
    for (int j = 0; j < 4; ++j) {
        const int d = threadIdx.x + j * 256;
        const size_t idx = (size_t)k * DIM + d;
        if (c > 0.f) centers[idx] = sums[idx] * inv;
        // else keep previous center
    }
}

__global__ __launch_bounds__(256) void copy_f32(const float* __restrict__ src,
                                                float* __restrict__ dst, int n)
{
    int i = blockIdx.x * blockDim.x + threadIdx.x;
    if (i < n) dst[i] = src[i];
}

// ---------------- launcher ----------------
extern "C" void kernel_launch(void* const* d_in, const int* in_sizes, int n_in,
                              void* d_out, int out_size, void* d_ws, size_t ws_size,
                              hipStream_t stream)
{
    const float* X = (const float*)d_in[0];   // [16384,1024]
    const float* W = (const float*)d_in[1];   // [512,512]
    float* out = (float*)d_out;               // [16384*512 out][512*1024 centers]

    char* ws = (char*)d_ws;
    size_t off = 0;
    __bf16* Xb      = (__bf16*)(ws + off); off += (size_t)M_ROWS * DIM * 2;    // 32 MB
    __bf16* Cb      = (__bf16*)(ws + off); off += (size_t)KCENT * DIM * 2;     // 1 MB
    float*  G       = (float*)(ws + off);  off += (size_t)M_ROWS * KCENT * 4;  // 32 MB
    float*  sums    = (float*)(ws + off);  off += (size_t)KCENT * DIM * 4;     // 2 MB
    float*  centers = (float*)(ws + off);  off += (size_t)KCENT * DIM * 4;     // 2 MB
    float*  cnorm   = (float*)(ws + off);  off += (size_t)KCENT * 4;
    float*  cnts    = (float*)(ws + off);  off += (size_t)KCENT * 4;
    int*    assign  = (int*)(ws + off);    off += (size_t)M_ROWS * 4;

    // setup
    {
        const int n = M_ROWS * DIM;
        convert_x_bf16<<<(n + 255) / 256, 256, 0, stream>>>(X, Xb, n);
        init_centers<<<KCENT, 256, 0, stream>>>(X, centers);
    }

    const dim3 gemm_grid(KCENT / 64, M_ROWS / 128);   // (8, 128)

    // Lloyd iterations (bf16 WMMA + async LDS staging for the distance gram)
    for (int it = 0; it < KM_IT; ++it) {
        prep_centers<<<KCENT, 256, 0, stream>>>(centers, Cb, cnorm);
        gemm_bf16_nt<<<gemm_grid, 256, 0, stream>>>(Xb, Cb, G, M_ROWS, KCENT, DIM);
        argmin_rows<<<M_ROWS / 8, 256, 0, stream>>>(G, cnorm, assign, KCENT);
        seg_sum<<<KCENT, 256, 0, stream>>>(X, assign, sums, cnts);
        update_centers<<<KCENT, 256, 0, stream>>>(sums, cnts, centers);
    }

    // final outputs in full fp32 on the matrix pipes
    gemm_f32_nt<<<gemm_grid, 256, 0, stream>>>(X, centers, G, M_ROWS, KCENT, DIM);
    gemm_f32_nt<<<gemm_grid, 256, 0, stream>>>(G, W, out, M_ROWS, KCENT, KCENT);

    // append centers after out
    {
        const int n = KCENT * DIM;
        copy_f32<<<(n + 255) / 256, 256, 0, stream>>>(centers, out + (size_t)M_ROWS * KCENT, n);
    }
}
